// SeqCondAttention_39968965657201
// MI455X (gfx1250) — matrix operand order repaired and checked by reading, not compiled, not run on hardware
//
#include <hip/hip_runtime.h>
#include <hip/hip_bf16.h>

// ---------------------------------------------------------------------------
// Problem constants (from the reference)
// ---------------------------------------------------------------------------
#define BB 2
#define LL 4096
#define DD 2048
#define KK 32
#define KQ 4
#define MM 8
#define HH 8
#define MEMC 256            // K*H
#define ZCH 288             // MEM + K
#define FLATC 2048          // K*H*M
#define MERGC 4128          // K + 2*FLAT
#define QCH 512             // KQ*H*M*2
#define Y3C 768             // K*H*3
#define NDEC 28             // K-4

typedef __attribute__((ext_vector_type(16))) _Float16 v16h;
typedef __attribute__((ext_vector_type(8)))  float    v8f;
typedef __attribute__((ext_vector_type(4)))  unsigned int v4u;
typedef __attribute__((ext_vector_type(8)))  int      v8i_t;
typedef __attribute__((ext_vector_type(4)))  int      v4i_t;

union FragH { v16h h; uint4 q[2]; };

// ---------------------------------------------------------------------------
// WMMA GEMM: C[M,N] = A[M,K] * B[K,N], fp32 in global, f16 WMMA in-core.
// Block = 128 threads = 4 waves; 128x64 C tile, 32-wide K steps.
// A tile (128x32 fp32, 16 KB) is DMA'd into LDS by the Tensor Data Mover
// (one tensor_load_to_lds per k-step, issued by wave 0, TENSORcnt-tracked).
// B tile is staged transposed + f16-converted by all threads.
// Each wave: 32-row strip -> 2 A-frags x 4 B-frags = 8 WMMAs per k-step.
// Requirements: Mrows % 128 == 0, Kdim % 32 == 0; N is bounds-checked.
// ---------------------------------------------------------------------------
#define TM 128
#define TN 64
#define TK 32

__global__ __launch_bounds__(128)
void wmma_gemm_f32f16(const float* __restrict__ A,
                      const float* __restrict__ Bm,
                      float* __restrict__ C,
                      int Mrows, int Ndim, int Kdim) {
    __shared__ float    As32[TM][TK];   // 16 KB, filled by TDM
    __shared__ _Float16 Bt[TN][TK];     // 4 KB, transposed f16

    const int tid    = threadIdx.x;
    const int wave   = tid >> 5;          // 0..3
    const int lane   = tid & 31;
    const int laneLo = lane & 15;
    const int laneHi = lane >> 4;
    const int row0   = blockIdx.y * TM;
    const int col0   = blockIdx.x * TN;

    v8f acc[2][4];
#pragma unroll
    for (int s = 0; s < 2; ++s)
#pragma unroll
        for (int j = 0; j < 4; ++j)
#pragma unroll
            for (int v = 0; v < 8; ++v) acc[s][j][v] = 0.0f;

    // B cooperative-load coordinates: 32x64 floats / 128 threads = 16 each
    const int bk = tid >> 2;             // k-row 0..31
    const int bn = (tid & 3) * 16;       // 16-col chunk
    const bool fullN = (col0 + TN) <= Ndim;

    const unsigned lds_a = (unsigned)(size_t)(&As32[0][0]);

    for (int k0 = 0; k0 < Kdim; k0 += TK) {
        // ---- Tensor Data Mover: DMA 128x32 fp32 A tile -> LDS (wave 0) ----
        if (wave == 0) {
            unsigned long long ga =
                (unsigned long long)(size_t)(A + (size_t)row0 * Kdim + k0);
            v4u g0;
            g0[0] = 1u;                                    // count=1 (valid D#)
            g0[1] = lds_a;                                 // lds_addr (bytes)
            g0[2] = (unsigned)(ga & 0xFFFFFFFFu);          // global_addr lo
            g0[3] = (unsigned)((ga >> 32) & 0x1FFFFFFu)    // global_addr hi
                    | (2u << 30);                          // type = 2 ("image")
            v8i_t g1;
            g1[0] = (int)(2u << 16);                       // data_size = 4 bytes
            g1[1] = (int)(((unsigned)TK & 0xFFFFu) << 16); // tensor_dim0 lo16
            g1[2] = (int)(((unsigned)TK >> 16) |
                          (((unsigned)TM & 0xFFFFu) << 16)); // dim0 hi / dim1 lo
            g1[3] = (int)(((unsigned)TM >> 16) |
                          ((unsigned)TK << 16));           // dim1 hi / tile_dim0=32
            g1[4] = (int)((unsigned)TM);                   // tile_dim1=128, tile_dim2=0
            g1[5] = (int)((unsigned)Kdim);                 // tensor_dim0_stride lo32
            g1[6] = 0;                                     // stride hi / dim1_stride lo
            g1[7] = 0;
            v4i_t gz4 = {0, 0, 0, 0};                      // 2D tensor: groups 2/3 zero
            v8i_t gz8 = {0, 0, 0, 0, 0, 0, 0, 0};
            __builtin_amdgcn_tensor_load_to_lds(g0, g1, gz4, gz4, gz8, 0);
        }

        // ---- stage B tile transposed: Bt[n][k], f16 ----
        const float* srcB = Bm + (size_t)(k0 + bk) * Ndim + col0 + bn;
        if (k0 + TK < Kdim)                                // -> global_prefetch_b8
            __builtin_prefetch(srcB + (size_t)TK * Ndim, 0, 3);
        if (fullN) {
#pragma unroll
            for (int i = 0; i < 16; i += 4) {
                float4 f = *(const float4*)(srcB + i);
                Bt[bn + i + 0][bk] = (_Float16)f.x;
                Bt[bn + i + 1][bk] = (_Float16)f.y;
                Bt[bn + i + 2][bk] = (_Float16)f.z;
                Bt[bn + i + 3][bk] = (_Float16)f.w;
            }
        } else {
            const float* brow = Bm + (size_t)(k0 + bk) * Ndim;
#pragma unroll
            for (int i = 0; i < 16; ++i) {
                int n  = col0 + bn + i;
                int nc = (n < Ndim) ? n : (Ndim - 1);      // clamped, in-bounds
                float v = brow[nc];
                Bt[bn + i][bk] = (_Float16)((n < Ndim) ? v : 0.0f);
            }
        }

        if (wave == 0) __builtin_amdgcn_s_wait_tensorcnt(0);
        __syncthreads();

        // ---- A fragments from fp32 LDS tile (ISA 7.12.2 16-bit A layout) ----
        // halves 0..7  -> K = e + 8*laneHi ; halves 8..15 -> K = 16+e+8*laneHi
        FragH a[2];
#pragma unroll
        for (int s = 0; s < 2; ++s) {
            const float* arow = &As32[wave * 32 + s * 16 + laneLo][0];
#pragma unroll
            for (int e = 0; e < 8; ++e) {
                a[s].h[e]     = (_Float16)arow[8 * laneHi + e];
                a[s].h[8 + e] = (_Float16)arow[16 + 8 * laneHi + e];
            }
        }
        // ---- 4 B fragments, each reused by 2 WMMAs ----
#pragma unroll
        for (int j = 0; j < 4; ++j) {
            FragH b;                       // lane = N, halves e<8 -> K = e+8*laneHi, e>=8 -> K = 16+(e-8)+8*laneHi
            const _Float16* brow = &Bt[j * 16 + laneLo][0];
            b.q[0] = *(const uint4*)(brow + 8 * laneHi);
            b.q[1] = *(const uint4*)(brow + 16 + 8 * laneHi);
            acc[0][j] = __builtin_amdgcn_wmma_f32_16x16x32_f16(
                false, a[0].h, false, b.h, (short)0, acc[0][j], false, false);
            acc[1][j] = __builtin_amdgcn_wmma_f32_16x16x32_f16(
                false, a[1].h, false, b.h, (short)0, acc[1][j], false, false);
        }
        __syncthreads();
    }

    // ---- epilogue: C/D layout -> VGPR v holds M = v + 8*laneHi, N = laneLo
#pragma unroll
    for (int s = 0; s < 2; ++s) {
#pragma unroll
        for (int j = 0; j < 4; ++j) {
            int c = col0 + j * 16 + laneLo;
            if (c < Ndim) {
#pragma unroll
                for (int v = 0; v < 8; ++v) {
                    int r = row0 + wave * 32 + s * 16 + 8 * laneHi + v;
                    C[(size_t)r * Ndim + c] = acc[s][j][v];
                }
            }
        }
    }
}

// ---------------------------------------------------------------------------
// Depthwise causal conv, window 4, 288 channels. thread = (b,l,c)
// ---------------------------------------------------------------------------
__global__ void conv_kernel(const float* __restrict__ zpre,
                            const float* __restrict__ ck,
                            float* __restrict__ zconv, int total) {
    int i = blockIdx.x * blockDim.x + threadIdx.x;
    if (i >= total) return;
    int c  = i % ZCH;
    int bl = i / ZCH;
    int l  = bl % LL;
    float acc = 0.0f;
#pragma unroll
    for (int w = 0; w < 4; ++w) {
        int ls = l - 3 + w;
        if (ls >= 0)
            acc += zpre[(size_t)(bl - l + ls) * ZCH + c] * ck[w * ZCH + c];
    }
    zconv[i] = acc;
}

// ---------------------------------------------------------------------------
// Gating + rotary feature map -> merged[b][l][4128]. thread = (b,l,k)
// ---------------------------------------------------------------------------
__global__ void merged_kernel(const float* __restrict__ zconv,
                              const float* __restrict__ theta,
                              const float* __restrict__ decay,
                              const float* __restrict__ anchor,
                              const float* __restrict__ sscale,
                              float* __restrict__ merged, int total) {
    int i = blockIdx.x * blockDim.x + threadIdx.x;
    if (i >= total) return;
    int k  = i % KK;
    int bl = i / KK;
    int l  = bl % LL;

    const float* zrow = zconv + (size_t)bl * ZCH;
    float s  = zrow[MEMC + k];
    float lp = fminf(fmaxf(sscale[k] * s, -20.0f), 20.0f);
    float lw;
    if (k < NDEC) {
        float sl = log1pf(expf(decay[k]));          // softplus
        lw = -sl * (float)(LL - 1 - l);
    } else {
        float sl = log1pf(expf(anchor[k - NDEC]));
        lw = -sl * (float)l;
    }
    float pw = expf(lp + lw);

    float* mrow = merged + (size_t)bl * MERGC;
    mrow[k] = pw;
#pragma unroll
    for (int h = 0; h < HH; ++h) {
        float kv = zrow[k * HH + h];
#pragma unroll
        for (int m = 0; m < MM; ++m) {
            float phi = kv * theta[(k * HH + h) * MM + m];
            int idx = (k * HH + h) * MM + m;
            mrow[KK + idx]         = pw * __cosf(phi);
            mrow[KK + FLATC + idx] = pw * __sinf(phi);
        }
    }
}

// ---------------------------------------------------------------------------
// In-place cumsum over L. thread = channel (coalesced 128B lines per wave).
// ---------------------------------------------------------------------------
__global__ void scan_kernel(float* __restrict__ merged) {
    int c = blockIdx.x * blockDim.x + threadIdx.x;
    int b = blockIdx.y;
    if (c >= MERGC) return;
    size_t base = (size_t)b * LL * MERGC + c;
    float acc = 0.0f;
    for (int l = 0; l < LL; ++l) {
        size_t idx = base + (size_t)l * MERGC;
        acc += merged[idx];
        merged[idx] = acc;
    }
}

// ---------------------------------------------------------------------------
// Readout: state = cs/den, combine with q, norm, W_re/W_im matmul, SwiGLU.
// thread = (b,l,k), writes 24 floats of y3.
// ---------------------------------------------------------------------------
__global__ void readout_kernel(const float* __restrict__ cs,
                               const float* __restrict__ qbuf,
                               const float* __restrict__ nscale,
                               const float* __restrict__ Wre,
                               const float* __restrict__ Wim,
                               float* __restrict__ y3, int total) {
    int i = blockIdx.x * blockDim.x + threadIdx.x;
    if (i >= total) return;
    int k  = i % KK;
    int bl = i / KK;

    const float* row = cs + (size_t)bl * MERGC;
    float inv = 1.0f / fmaxf(row[k], 1e-4f);
    const float* q = qbuf + (size_t)bl * QCH + (size_t)(k >> 3) * (HH * MM * 2);

    float ore[HH], oim[HH];
#pragma unroll
    for (int h = 0; h < HH; ++h) {
        const float* nr = row + KK + (k * HH + h) * MM;
        const float* ni = nr + FLATC;
        const float* qh = q + h * MM * 2;
        float re = 0.0f, im = 0.0f;
#pragma unroll
        for (int m = 0; m < MM; ++m) {
            float sre = nr[m] * inv, sim = ni[m] * inv;
            float qr = qh[2 * m], qi = qh[2 * m + 1];
            re += sre * qr + sim * qi;
            im += sim * qr - sre * qi;
        }
        float ns = nscale[k * HH + h];
        ore[h] = re * ns;
        oim[h] = im * ns;
    }

    float* out = y3 + (size_t)bl * Y3C + k * 24;
    const float* wr = Wre + (size_t)k * HH * 48;
    const float* wi = Wim + (size_t)k * HH * 48;
#pragma unroll
    for (int n = 0; n < 24; ++n) {
        float yv = 0.0f, yg = 0.0f;
#pragma unroll
        for (int h = 0; h < HH; ++h) {
            yv += ore[h] * wr[h * 48 + n]      + oim[h] * wi[h * 48 + n];
            yg += ore[h] * wr[h * 48 + 24 + n] + oim[h] * wi[h * 48 + 24 + n];
        }
        float sg = 1.0f / (1.0f + expf(-yg));
        out[n] = yv * (yg * sg);               // y_val * silu(y_gate)
    }
}

// ---------------------------------------------------------------------------
// Host-side orchestration
// ---------------------------------------------------------------------------
extern "C" void kernel_launch(void* const* d_in, const int* in_sizes, int n_in,
                              void* d_out, int out_size, void* d_ws, size_t ws_size,
                              hipStream_t stream) {
    const float* x      = (const float*)d_in[0];
    const float* W_mem  = (const float*)d_in[1];
    const float* conv_k = (const float*)d_in[2];
    const float* W_q    = (const float*)d_in[3];
    const float* theta  = (const float*)d_in[4];
    const float* decay  = (const float*)d_in[5];
    const float* anchor = (const float*)d_in[6];
    const float* sscale = (const float*)d_in[7];
    const float* W_re   = (const float*)d_in[8];
    const float* W_im   = (const float*)d_in[9];
    const float* nscale = (const float*)d_in[10];
    const float* W_out  = (const float*)d_in[11];
    float*       out    = (float*)d_out;

    const size_t NBL = (size_t)BB * LL;      // 8192 rows
    float* zpre   = (float*)d_ws;
    float* zconv  = zpre   + NBL * ZCH;
    float* qbuf   = zconv  + NBL * ZCH;
    float* merged = qbuf   + NBL * QCH;
    float* y3     = merged + NBL * MERGC;
    (void)ws_size; (void)in_sizes; (void)n_in; (void)out_size;

    const int Mrows = (int)NBL;

    // GEMM1: zpre = x @ W_mem    (8192x2048 x 2048x288)
    wmma_gemm_f32f16<<<dim3((ZCH + TN - 1) / TN, Mrows / TM), 128, 0, stream>>>(
        x, W_mem, zpre, Mrows, ZCH, DD);
    // GEMM2: qbuf = x @ W_q      (8192x2048 x 2048x512)
    wmma_gemm_f32f16<<<dim3(QCH / TN, Mrows / TM), 128, 0, stream>>>(
        x, W_q, qbuf, Mrows, QCH, DD);

    // depthwise causal conv
    {
        int total = (int)(NBL * ZCH);
        conv_kernel<<<(total + 255) / 256, 256, 0, stream>>>(zpre, conv_k, zconv, total);
    }
    // feature map -> merged
    {
        int total = (int)(NBL * KK);
        merged_kernel<<<(total + 255) / 256, 256, 0, stream>>>(
            zconv, theta, decay, anchor, sscale, merged, total);
    }
    // in-place cumsum over L
    scan_kernel<<<dim3((MERGC + 255) / 256, BB), 256, 0, stream>>>(merged);

    // readout + tiny matmuls + SwiGLU -> y3
    {
        int total = (int)(NBL * KK);
        readout_kernel<<<(total + 255) / 256, 256, 0, stream>>>(
            merged, qbuf, nscale, W_re, W_im, y3, total);
    }

    // GEMM3: out = y3 @ W_out    (8192x768 x 768x2048)
    wmma_gemm_f32f16<<<dim3(DD / TN, Mrows / TM), 128, 0, stream>>>(
        y3, W_out, out, Mrows, DD, Y3C);
}
